// SelfAttention_31568009626123
// MI455X (gfx1250) — matrix operand order repaired and checked by reading, not compiled, not run on hardware
//
#include <hip/hip_runtime.h>
#include <hip/hip_bf16.h>

typedef __attribute__((ext_vector_type(16))) _Float16 v16h;
typedef __attribute__((ext_vector_type(8)))  _Float16 v8h;
typedef __attribute__((ext_vector_type(4)))  _Float16 v4h;
typedef __attribute__((ext_vector_type(8)))  float    v8f;

#define NTOK 4096
#define DIM  1024

// ---------------------------------------------------------------------------
// WMMA helpers (CDNA5 16x16x32 f16, f32 accumulate)
// ---------------------------------------------------------------------------
__device__ __forceinline__ v8f wmma_f16(v16h a, v16h b, v8f c) {
    // (neg_a, A, neg_b, B, c_mod, C, reuse_a, reuse_b)
    return __builtin_amdgcn_wmma_f32_16x16x32_f16(false, a, false, b,
                                                  (short)0, c, false, false);
}

// Build a 16-half fragment from two contiguous 8-half (16B) chunks.
// A-fragment (16x32, row-major src): p1 = p0 + 16 elements.
// B-fragment (32x16 from Bt[n][k] src): p1 = p0 + 16 elements.
__device__ __forceinline__ v16h ld_frag2(const _Float16* p0, const _Float16* p1) {
    v8h lo = *(const v8h*)p0;
    v8h hi = *(const v8h*)p1;
    v16h r;
#pragma unroll
    for (int i = 0; i < 8; ++i) { r[i] = lo[i]; r[i + 8] = hi[i]; }
    return r;
}

// ---------------------------------------------------------------------------
// Stage 1: x f32 -> f16 (same layout)
// ---------------------------------------------------------------------------
__global__ __launch_bounds__(256) void conv_x_kernel(const float* __restrict__ in,
                                                     _Float16* __restrict__ out) {
    int i = blockIdx.x * 256 + threadIdx.x;           // one float4 per thread
    float4 v = ((const float4*)in)[i];
    v4h o;
    o[0] = (_Float16)v.x; o[1] = (_Float16)v.y;
    o[2] = (_Float16)v.z; o[3] = (_Float16)v.w;
    ((v4h*)out)[i] = o;
}

// ---------------------------------------------------------------------------
// Stage 2: W [K][N] f32 -> Wt [N][K] f16 (transpose through LDS)
// ---------------------------------------------------------------------------
__global__ __launch_bounds__(256) void conv_wt_kernel(const float* __restrict__ wq,
                                                      const float* __restrict__ wk,
                                                      const float* __restrict__ wv,
                                                      _Float16* __restrict__ wt_all) {
    __shared__ float tile[32][33];
    int z = blockIdx.z;
    const float* src = (z == 0) ? wq : ((z == 1) ? wk : wv);
    _Float16* dst = wt_all + (size_t)z * DIM * DIM;
    int nb = blockIdx.x * 32;   // output-row block (N)
    int kb = blockIdx.y * 32;   // K block
    int t = threadIdx.x;

    // coalesced read: thread t reads W[kb + t/8][nb + (t%8)*4 .. +3]
    int k  = t >> 3;
    int n4 = (t & 7) * 4;
    float4 v = *(const float4*)(src + (size_t)(kb + k) * DIM + nb + n4);
    tile[k][n4 + 0] = v.x; tile[k][n4 + 1] = v.y;
    tile[k][n4 + 2] = v.z; tile[k][n4 + 3] = v.w;
    __syncthreads();

    // coalesced write: thread t writes Wt[nb + t/8][kb + (t%8)*4 .. +3]
    int n  = t >> 3;
    int k4 = (t & 7) * 4;
    v4h o;
#pragma unroll
    for (int i = 0; i < 4; ++i) o[i] = (_Float16)tile[k4 + i][n];
    *(v4h*)(dst + (size_t)(nb + n) * DIM + kb + k4) = o;
}

// ---------------------------------------------------------------------------
// Stage 3: fused QKV projection GEMM.
// grid (NTOK/32, 2, 3), block 256 (8 waves). Wave computes a 32x64 tile
// (two 16-row A fragments share every B fragment -> 1.5 b128 loads per WMMA).
// z=0: Q16 = (x@Wq + bq)/32 (scale folded)     -> q16 [tok][feat] f16
// z=1: K16 =  x@Wk + bk                        -> k16 [tok][feat] f16
// z=2: Vt16 = (x@Wv + bv)^T                    -> vt16 [feat][tok] f16
// ---------------------------------------------------------------------------
__global__ __launch_bounds__(256) void qkv_gemm_kernel(
        const _Float16* __restrict__ x16, const _Float16* __restrict__ wt_all,
        const float* __restrict__ bq, const float* __restrict__ bk,
        const float* __restrict__ bv,
        _Float16* __restrict__ q16, _Float16* __restrict__ k16,
        _Float16* __restrict__ vt16) {
    int z = blockIdx.z;
    const _Float16* wt = wt_all + (size_t)z * DIM * DIM;
    const float* bias = (z == 0) ? bq : ((z == 1) ? bk : bv);

    int lane = threadIdx.x & 31;
    int wave = threadIdx.x >> 5;
    int mbase = blockIdx.x * 32;
    int nbase = blockIdx.y * 512 + wave * 64;
    int n0 = lane & 15;         // A: row m within tile; B: col n within tile
    int hi = lane >> 4;         // lane half selects K sub-chunk / M sub-half

    v8f acc[2][4] = {};
    const _Float16* arow0 = x16 + (size_t)(mbase + n0) * DIM;
    const _Float16* arow1 = x16 + (size_t)(mbase + 16 + n0) * DIM;

#pragma unroll 2
    for (int ks = 0; ks < DIM; ks += 32) {
        const _Float16* a0p = arow0 + ks + hi * 8;
        const _Float16* a1p = arow1 + ks + hi * 8;
        v16h a0 = ld_frag2(a0p, a0p + 16);
        v16h a1 = ld_frag2(a1p, a1p + 16);
#pragma unroll
        for (int t = 0; t < 4; ++t) {
            const _Float16* bp =
                wt + (size_t)(nbase + t * 16 + n0) * DIM + ks + hi * 8;
            v16h b = ld_frag2(bp, bp + 16);
            acc[0][t] = wmma_f16(a0, b, acc[0][t]);
            acc[1][t] = wmma_f16(a1, b, acc[1][t]);
        }
    }

    if (z < 2) {
        _Float16* dst = (z == 0) ? q16 : k16;
        float scale = (z == 0) ? 0.03125f : 1.0f;   // 1/sqrt(1024) folded into Q
#pragma unroll
        for (int t = 0; t < 4; ++t) {
            float bb = bias[nbase + t * 16 + n0];
#pragma unroll
            for (int g = 0; g < 2; ++g) {
#pragma unroll
                for (int r = 0; r < 8; ++r) {
                    int m = mbase + g * 16 + hi * 8 + r;
                    dst[(size_t)m * DIM + nbase + t * 16 + n0] =
                        (_Float16)((acc[g][t][r] + bb) * scale);
                }
            }
        }
    } else {
        // transposed store: per lane, rows r=0..7 are token-contiguous -> one b128
#pragma unroll
        for (int t = 0; t < 4; ++t) {
            float bb = bias[nbase + t * 16 + n0];
#pragma unroll
            for (int g = 0; g < 2; ++g) {
                v8h pack;
#pragma unroll
                for (int r = 0; r < 8; ++r) pack[r] = (_Float16)(acc[g][t][r] + bb);
                *(v8h*)(vt16 + (size_t)(nbase + t * 16 + n0) * NTOK + mbase +
                        g * 16 + hi * 8) = pack;
            }
        }
    }
}

// ---------------------------------------------------------------------------
// Stage 4: causal flash attention.
// grid (NTOK/32), block 256 = 8 waves. Block owns 32 query rows (2 groups
// of 16), so every K-tile / V-tile B fragment feeds two WMMAs and the serial
// softmax + barriers amortize over 2x the matrix work.
// Wave w owns feature slab [w*128, w*128+128): O accumulator = 2 x 8 x v8f.
// Score contraction over 1024 features is split-K across the 8 waves and
// reduced through LDS; softmax (online) by threads 0..31; P staged in LDS.
// ---------------------------------------------------------------------------
__global__ __launch_bounds__(256) void attn_kernel(
        const _Float16* __restrict__ q16, const _Float16* __restrict__ k16,
        const _Float16* __restrict__ vt16, float* __restrict__ out) {
    __shared__ float s_partial[8][32][32];            // split-K score partials
    __shared__ float s_sum[32][32];                   // reduced scores
    __shared__ alignas(16) _Float16 s_p[32][32];      // softmax probs (A frag src)
    __shared__ float s_alpha[32];                     // per-row rescale
    __shared__ float s_linv[32];                      // 1/l final

    int tid = threadIdx.x;
    int lane = tid & 31;
    int wave = tid >> 5;
    int n0 = lane & 15;
    int hi = lane >> 4;
    int qbase = blockIdx.x * 32;
    int fs = wave * 128;                              // this wave's feature slab

    v8f acc[2][8] = {};
    float row_m = -3.0e38f, row_l = 0.0f;             // live in threads 0..31

    // register-cache this wave's Q fragments (2 groups x 16 rows x 128 feats)
    v16h qa[2][4];
#pragma unroll
    for (int g = 0; g < 2; ++g) {
        const _Float16* qrow = q16 + (size_t)(qbase + g * 16 + n0) * DIM;
#pragma unroll
        for (int kk = 0; kk < 4; ++kk) {
            const _Float16* qp = qrow + fs + kk * 32 + hi * 8;
            qa[g][kk] = ld_frag2(qp, qp + 16);
        }
    }

    int nkb = (qbase + 32) >> 5;                      // 32-key blocks, causal bound
    for (int kb = 0; kb < nkb; ++kb) {
        int kbase = kb * 32;

        // ---- score partials over this wave's 128 features ----
        v8f sg[2][2] = {};
#pragma unroll
        for (int kk = 0; kk < 4; ++kk) {
            int ks = fs + kk * 32;
            const _Float16* b0p = k16 + (size_t)(kbase + n0) * DIM + ks + hi * 8;
            const _Float16* b1p =
                k16 + (size_t)(kbase + 16 + n0) * DIM + ks + hi * 8;
            v16h b0 = ld_frag2(b0p, b0p + 16);
            v16h b1 = ld_frag2(b1p, b1p + 16);
#pragma unroll
            for (int g = 0; g < 2; ++g) {
                sg[g][0] = wmma_f16(qa[g][kk], b0, sg[g][0]);
                sg[g][1] = wmma_f16(qa[g][kk], b1, sg[g][1]);
            }
        }
        if (kb + 1 < nkb) {                           // warm L2/WGP$ for next block
            __builtin_prefetch(k16 + (size_t)(kbase + 32 + n0) * DIM + fs, 0, 0);
            __builtin_prefetch(vt16 + (size_t)(fs + n0) * NTOK + kbase + 32, 0, 0);
        }
#pragma unroll
        for (int g = 0; g < 2; ++g) {
#pragma unroll
            for (int r = 0; r < 8; ++r) {
                s_partial[wave][g * 16 + r + hi * 8][n0]      = sg[g][0][r];
                s_partial[wave][g * 16 + r + hi * 8][16 + n0] = sg[g][1][r];
            }
        }
        __syncthreads();

        // ---- reduce 8 partials (1024 scores, 4 per thread) ----
        for (int i = tid; i < 1024; i += 256) {
            int rr = i >> 5, cc = i & 31;
            float v = 0.0f;
#pragma unroll
            for (int w = 0; w < 8; ++w) v += s_partial[w][rr][cc];
            s_sum[rr][cc] = v;
        }
        __syncthreads();

        // ---- online softmax, one row per thread (0..31), causal mask ----
        if (tid < 32) {
            int qi = qbase + tid;
            float m_new = row_m;
#pragma unroll
            for (int c = 0; c < 32; ++c) {
                float s = (kbase + c <= qi) ? s_sum[tid][c] : -3.0e38f;
                m_new = fmaxf(m_new, s);
            }
            float alpha = __expf(row_m - m_new);
            float lsum = 0.0f;
#pragma unroll
            for (int c = 0; c < 32; ++c) {
                float p;
                if (kbase + c <= qi) p = __expf(s_sum[tid][c] - m_new);
                else                 p = 0.0f;
                s_p[tid][c] = (_Float16)p;
                lsum += p;
            }
            row_l = row_l * alpha + lsum;
            row_m = m_new;
            s_alpha[tid] = alpha;
        }
        __syncthreads();

        // ---- rescale O and accumulate P @ V over this wave's feature slab ----
        float al[2][8];
        v16h pa[2];
#pragma unroll
        for (int g = 0; g < 2; ++g) {
#pragma unroll
            for (int r = 0; r < 8; ++r) al[g][r] = s_alpha[g * 16 + r + hi * 8];
            const _Float16* pp = &s_p[g * 16 + n0][0] + hi * 8;
            pa[g] = ld_frag2(pp, pp + 16);
        }
#pragma unroll
        for (int t = 0; t < 8; ++t) {
            const _Float16* vp =
                vt16 + (size_t)(fs + t * 16 + n0) * NTOK + kbase + hi * 8;
            v16h b = ld_frag2(vp, vp + 16);
#pragma unroll
            for (int g = 0; g < 2; ++g) {
#pragma unroll
                for (int r = 0; r < 8; ++r) acc[g][t][r] *= al[g][r];
                acc[g][t] = wmma_f16(pa[g], b, acc[g][t]);
            }
        }
    }

    // ---- normalize by 1/l and store f32 output ----
    if (tid < 32) s_linv[tid] = 1.0f / row_l;
    __syncthreads();
    float li[2][8];
#pragma unroll
    for (int g = 0; g < 2; ++g)
#pragma unroll
        for (int r = 0; r < 8; ++r) li[g][r] = s_linv[g * 16 + r + hi * 8];
#pragma unroll
    for (int t = 0; t < 8; ++t) {
#pragma unroll
        for (int g = 0; g < 2; ++g) {
#pragma unroll
            for (int r = 0; r < 8; ++r) {
                out[(size_t)(qbase + g * 16 + hi * 8 + r) * DIM + fs + t * 16 +
                    n0] = acc[g][t][r] * li[g][r];
            }
        }
    }
}

// ---------------------------------------------------------------------------
// Launcher
// ---------------------------------------------------------------------------
extern "C" void kernel_launch(void* const* d_in, const int* in_sizes, int n_in,
                              void* d_out, int out_size, void* d_ws, size_t ws_size,
                              hipStream_t stream) {
    (void)in_sizes; (void)n_in; (void)out_size; (void)ws_size;
    const float* x  = (const float*)d_in[0];
    const float* Wq = (const float*)d_in[1];
    const float* bq = (const float*)d_in[2];
    const float* Wk = (const float*)d_in[3];
    const float* bk = (const float*)d_in[4];
    const float* Wv = (const float*)d_in[5];
    const float* bv = (const float*)d_in[6];
    float* out = (float*)d_out;

    // workspace layout (f16): x16 | q16 | k16 | vt16 | wt(3x) = 38 MB
    _Float16* ws   = (_Float16*)d_ws;
    _Float16* x16  = ws;
    _Float16* q16  = x16 + (size_t)NTOK * DIM;
    _Float16* k16  = q16 + (size_t)NTOK * DIM;
    _Float16* vt16 = k16 + (size_t)NTOK * DIM;
    _Float16* wt16 = vt16 + (size_t)NTOK * DIM;

    conv_x_kernel<<<(NTOK * DIM) / (256 * 4), 256, 0, stream>>>(x, x16);
    conv_wt_kernel<<<dim3(32, 32, 3), 256, 0, stream>>>(Wq, Wk, Wv, wt16);
    qkv_gemm_kernel<<<dim3(NTOK / 32, 2, 3), 256, 0, stream>>>(
        x16, wt16, bq, bk, bv, q16, k16, vt16);
    attn_kernel<<<NTOK / 32, 256, 0, stream>>>(q16, k16, vt16, out);
}
